// PSMCosineLayer_41858751267257
// MI455X (gfx1250) — compile-verified, matrix operand order checked by reading, not run on hardware
//
#include <hip/hip_runtime.h>
#include <hip/hip_bf16.h>

// Problem constants (from reference): B,H,W,C = 4,72,240,320 ; D = 48
#define NB 4
#define NH 72
#define NW 240
#define NC 320
#define ND 48
#define WTILES (NW / 16)            // 15 w-tiles per (b,h) row
#define NTILES (NB * NH * WTILES)   // 4320 total w-tiles

typedef __attribute__((ext_vector_type(16))) __bf16 v16bf;
typedef __attribute__((ext_vector_type(8)))  float  v8f;
typedef __attribute__((ext_vector_type(4)))  float  v4f;

// Load one 16-lane x K=32 bf16 WMMA fragment (A or B operand) for this lane.
// Per-lane element mapping for 16-bit operands (wave32 ISA layout):
//   slots 0..7  <- K = half*8 + 0..7        (VGPR0..3, K pairs lo/hi)
//   slots 8..15 <- K = 16 + half*8 + 0..7   (VGPR4..7)
__device__ __forceinline__ v16bf load_frag_bf16(const float* __restrict__ row, int cA)
{
    v4f x0 = *(const v4f*)(row + cA);
    v4f x1 = *(const v4f*)(row + cA + 4);
    v4f x2 = *(const v4f*)(row + cA + 16);
    v4f x3 = *(const v4f*)(row + cA + 20);
    float f[16];
    *(v4f*)&f[0]  = x0; *(v4f*)&f[4]  = x1;
    *(v4f*)&f[8]  = x2; *(v4f*)&f[12] = x3;
    v16bf r;
#pragma unroll
    for (int e = 0; e < 16; ++e) r[e] = (__bf16)f[e];
    return r;
}

// One wave owns one 16-wide w-tile of one (b,h) row and computes the full
// 16 x 48 disparity band as 4 independent 16x16 (w,j) GEMM tiles
// (j0 = w0-48 .. w0), sharing a single A fragment per K-chunk.
// Rows with j<0 are loaded clamped (garbage) and fixed to 0.0 at store time
// (predicate i > w), keeping the K-loop control flow fully uniform so EXEC
// is all-1s for every V_WMMA as the ISA requires.
__global__ __launch_bounds__(128) void cost_volume_wmma_kernel(
    const float* __restrict__ left,
    const float* __restrict__ right,
    float* __restrict__ out)
{
    const int lane = threadIdx.x & 31;          // wave32
    const int wave = threadIdx.x >> 5;          // 0..3
    const int tile = blockIdx.x * 4 + wave;     // 0 .. NTILES-1

    const int wt = tile % WTILES;
    const int bh = tile / WTILES;               // b*H + h
    const int w0 = wt * 16;

    const int row16 = lane & 15;                // matrix row handled by this lane
    const int half  = lane >> 4;                // K-half select

    const float* __restrict__ Lrow = left + ((long)bh * NW + w0 + row16) * NC;

    const float* Rrow[4];
#pragma unroll
    for (int jt = 0; jt < 4; ++jt) {
        int jrow = w0 - 48 + jt * 16 + row16;   // right (B) row for this j-tile
        if (jrow < 0) jrow = 0;                 // clamp; fixed up at store
        Rrow[jt] = right + ((long)bh * NW + jrow) * NC;
    }

    v8f acc[4] = {};

#pragma unroll
    for (int c0 = 0; c0 < NC; c0 += 32) {
        const int cA = c0 + half * 8;
        const v16bf A = load_frag_bf16(Lrow, cA);   // shared across 4 WMMAs
#pragma unroll
        for (int jt = 0; jt < 4; ++jt) {
            const v16bf Bv = load_frag_bf16(Rrow[jt], cA);
            // (neg_a, A, neg_b, B, c_mod, C, reuse_a, reuse_b)
            acc[jt] = __builtin_amdgcn_wmma_f32_16x16x32_bf16(
                false, A, false, Bv, (short)0, acc[jt], false, false);
        }
    }

    // D-matrix layout (32-bit 16x16): VGPR r, lanes 0-15 -> M=r, N=lane;
    // lanes 16-31 -> M=r+8, N=lane-16.
    //   w = w0 + m ; j = j0 + n ; i = w - j ; store iff 0 <= i < 48.
    // Across the 4 j-tiles every valid (w,i) appears exactly once.
    // i > w  <=>  j < 0  -> reference zero-pads: store 0.0f there.
    const float scale = 1.0f / (float)NC;
#pragma unroll
    for (int jt = 0; jt < 4; ++jt) {
        const int dbase = 48 - jt * 16;         // w0 - j0 for this j-tile
#pragma unroll
        for (int r = 0; r < 8; ++r) {
            const int m = r + half * 8;
            const int n = row16;
            const int i = dbase + m - n;
            if (i >= 0 && i < ND) {
                const int w = w0 + m;
                const float val = (i <= w) ? acc[jt][r] * scale : 0.0f;
                out[((long)bh * NW + w) * ND + i] = val;
            }
        }
    }
}

extern "C" void kernel_launch(void* const* d_in, const int* in_sizes, int n_in,
                              void* d_out, int out_size, void* d_ws, size_t ws_size,
                              hipStream_t stream) {
    const float* left  = (const float*)d_in[0];
    const float* right = (const float*)d_in[1];
    float* out = (float*)d_out;

    dim3 grid(NTILES / 4);   // 1080 workgroups, 4 tiles (one per wave) each
    dim3 block(128);         // 4 wave32 waves
    hipLaunchKernelGGL(cost_volume_wmma_kernel, grid, block, 0, stream,
                       left, right, out);
}
